// MPNNLayer_18854906429487
// MI455X (gfx1250) — matrix operand order repaired and checked by reading, not compiled, not run on hardware
//
#include <hip/hip_runtime.h>
#include <hip/hip_bf16.h>

#define NN 50000
#define NE 800000
#define ND 128
#define ED 64
#define HID 256

typedef _Float16 v16h __attribute__((ext_vector_type(16)));
typedef _Float16 v8h  __attribute__((ext_vector_type(8)));
typedef _Float16 v4h  __attribute__((ext_vector_type(4)));
typedef float    v8f  __attribute__((ext_vector_type(8)));

// A-fragment for v_wmma_f32_16x16x32_f16 (wave32):
// lane L holds row M = L%16; vector elems 0..7  = K0+ (L/16)*8 + {0..7}
//                            vector elems 8..15 = K0+16+(L/16)*8 + {0..7}
// -> two contiguous 8-half (16B) runs from a row-major f16 tile.
__device__ __forceinline__ v16h afrag(const _Float16* p) {
  v8h lo = *(const v8h*)(p);
  v8h hi = *(const v8h*)(p + 16);
  v16h a;
#pragma unroll
  for (int i = 0; i < 8; ++i) { a[i] = lo[i]; a[i + 8] = hi[i]; }
  return a;
}

__device__ __forceinline__ v8f wmma16(v16h a, v16h b, v8f c) {
  return __builtin_amdgcn_wmma_f32_16x16x32_f16(false, a, false, b, (short)0, c,
                                                false, false);
}

// ---------------- workspace prep ----------------

__global__ void k_zero(int4* __restrict__ p, int n4) {
  int i = blockIdx.x * blockDim.x + threadIdx.x;
  if (i < n4) p[i] = make_int4(0, 0, 0, 0);
}

// Pack row-major f32 W[K][ldN] into WMMA B-fragment order, f16:
// out[((nT*nK + kT)*32 + lane)*16 + i] = W[kT*32 + (lane/16)*8 + (i%8) + (i/8)*16][nT*16 + lane%16]
__global__ void k_pack(const float* __restrict__ W, _Float16* __restrict__ out,
                       int nK, int ldN, int total) {
  int idx = blockIdx.x * blockDim.x + threadIdx.x;
  if (idx >= total) return;
  int i    = idx & 15;
  int lane = (idx >> 4) & 31;
  int t    = idx >> 9;
  int kT   = t % nK;
  int nT   = t / nK;
  int K = kT * 32 + (lane >> 4) * 8 + (i & 7) + ((i >> 3) << 4);
  int N = nT * 16 + (lane & 15);
  out[idx] = (_Float16)W[(size_t)K * ldN + N];
}

__global__ void k_deg(const int* __restrict__ dst, int* __restrict__ deg) {
  int e = blockIdx.x * blockDim.x + threadIdx.x;
  if (e < NE) atomicAdd(&deg[dst[e]], 1);
}

// ---------------- edge MLP + scatter ----------------
// One block (4 waves) per 64-edge tile (4 M-tiles of 16) to amortize B traffic.
__global__ __launch_bounds__(128) void k_edge(
    const float* __restrict__ nodeF, const float* __restrict__ edgeF,
    const int* __restrict__ src, const int* __restrict__ dst,
    const _Float16* __restrict__ pW1, const float* __restrict__ b1,
    const _Float16* __restrict__ pW2, const float* __restrict__ b2,
    float* __restrict__ esum) {
  __shared__ alignas(16) _Float16 A[64 * 328];  // 64 x 320 f16, stride 328 (656B, 16B-mult)
  __shared__ alignas(16) _Float16 H[64 * 264];  // 64 x 256 f16, stride 264 (528B, 16B-mult)
  __shared__ int sSrc[64];
  __shared__ int sDst[64];

  const int tid   = threadIdx.x;
  const int lane  = tid & 31;
  const int wave  = tid >> 5;
  const int lrow  = lane & 15;
  const int lhalf = lane >> 4;
  const int e0    = blockIdx.x * 64;

  if (tid < 64) { sSrc[tid] = src[e0 + tid]; sDst[tid] = dst[e0 + tid]; }
  __syncthreads();

  // Gather [src_h(128) | e(64) | dst_h(128)] -> f16 LDS tile (float4 loads)
  for (int v = tid; v < 64 * 80; v += 128) {
    int r = v / 80;
    int c = (v % 80) * 4;
    const float* p;
    if (c < 128)      p = nodeF + (size_t)sSrc[r] * ND + c;
    else if (c < 192) p = edgeF + (size_t)(e0 + r) * ED + (c - 128);
    else              p = nodeF + (size_t)sDst[r] * ND + (c - 192);
    float4 d = *(const float4*)p;
    v4h h4 = { (_Float16)d.x, (_Float16)d.y, (_Float16)d.z, (_Float16)d.w };
    *(v4h*)&A[r * 328 + c] = h4;
  }
  __syncthreads();

  // GEMM1: [64x320] @ [320x256]; wave owns 4 N-tiles, all 4 M-tiles; 10 K-steps.
  // Each B fragment load feeds 4 WMMAs (reused across M-tiles).
  v8f acc[4][4] = {};
  for (int k = 0; k < 10; ++k) {
    v16h a[4];
#pragma unroll
    for (int m = 0; m < 4; ++m)
      a[m] = afrag(&A[(m * 16 + lrow) * 328 + k * 32 + lhalf * 8]);
#pragma unroll
    for (int n = 0; n < 4; ++n) {
      int nT = wave * 4 + n;
      v16h b = *(const v16h*)&pW1[(size_t)((nT * 10 + k) * 32 + lane) * 16];
#pragma unroll
      for (int m = 0; m < 4; ++m)
        acc[m][n] = wmma16(a[m], b, acc[m][n]);
    }
  }
  // bias + ReLU -> f16 hidden tile in LDS
#pragma unroll
  for (int n = 0; n < 4; ++n) {
    int ncol = (wave * 4 + n) * 16 + lrow;
    float bias = b1[ncol];
#pragma unroll
    for (int m = 0; m < 4; ++m) {
#pragma unroll
      for (int r = 0; r < 8; ++r) {
        float h = acc[m][n][r] + bias;
        H[(m * 16 + r + lhalf * 8) * 264 + ncol] = (_Float16)(h > 0.f ? h : 0.f);
      }
    }
  }
  __syncthreads();

  // GEMM2: [64x256] @ [256x64]; wave owns 1 N-tile, all 4 M-tiles; 8 K-steps.
  v8f c2[4] = {};
  for (int k = 0; k < 8; ++k) {
    v16h b = *(const v16h*)&pW2[(size_t)((wave * 8 + k) * 32 + lane) * 16];
#pragma unroll
    for (int m = 0; m < 4; ++m) {
      v16h a = afrag(&H[(m * 16 + lrow) * 264 + k * 32 + lhalf * 8]);
      c2[m] = wmma16(a, b, c2[m]);
    }
  }
  int ncol = wave * 16 + lrow;
  float bias = b2[ncol];
#pragma unroll
  for (int m = 0; m < 4; ++m) {
#pragma unroll
    for (int r = 0; r < 8; ++r) {
      int row = m * 16 + r + lhalf * 8;  // edge row within 64-edge tile
      atomicAdd(&esum[(size_t)sDst[row] * ED + ncol], c2[m][r] + bias);
    }
  }
}

// ---------------- node MLP ----------------
// One block (4 waves) per 16-node tile.
__global__ __launch_bounds__(128) void k_node(
    const float* __restrict__ nodeF, const float* __restrict__ esum,
    const int* __restrict__ deg,
    const _Float16* __restrict__ pW1, const float* __restrict__ b1,
    const _Float16* __restrict__ pW2, const float* __restrict__ b2,
    float* __restrict__ out) {
  __shared__ alignas(16) _Float16 A[16 * 200];  // 16 x 192, stride 200 (400B, 16B-mult)
  __shared__ alignas(16) _Float16 H[16 * 264];
  __shared__ int sDeg[16];

  const int tid   = threadIdx.x;
  const int lane  = tid & 31;
  const int wave  = tid >> 5;
  const int lrow  = lane & 15;
  const int lhalf = lane >> 4;
  const int n0    = blockIdx.x * 16;

  if (tid < 16) sDeg[tid] = deg[n0 + tid];
  for (int v = tid; v < 16 * 48; v += 128) {
    int r = v / 48;
    int c = (v % 48) * 4;
    const float* p = (c < 128) ? nodeF + (size_t)(n0 + r) * ND + c
                               : esum + (size_t)(n0 + r) * ED + (c - 128);
    float4 d = *(const float4*)p;
    v4h h4 = { (_Float16)d.x, (_Float16)d.y, (_Float16)d.z, (_Float16)d.w };
    *(v4h*)&A[r * 200 + c] = h4;
  }
  __syncthreads();

  // GEMM1: [16x192] @ [192x256], 4 N-tiles/wave, 6 K-steps
  v8f acc[4] = {};
  for (int k = 0; k < 6; ++k) {
    v16h a = afrag(&A[lrow * 200 + k * 32 + lhalf * 8]);
#pragma unroll
    for (int n = 0; n < 4; ++n) {
      int nT = wave * 4 + n;
      v16h b = *(const v16h*)&pW1[(size_t)((nT * 6 + k) * 32 + lane) * 16];
      acc[n] = wmma16(a, b, acc[n]);
    }
  }
#pragma unroll
  for (int n = 0; n < 4; ++n) {
    int ncol = (wave * 4 + n) * 16 + lrow;
    float bias = b1[ncol];
#pragma unroll
    for (int r = 0; r < 8; ++r) {
      float h = acc[n][r] + bias;
      H[(r + lhalf * 8) * 264 + ncol] = (_Float16)(h > 0.f ? h : 0.f);
    }
  }
  __syncthreads();

  // GEMM2: [16x256] @ [256x128], 2 N-tiles/wave, 8 K-steps
  v8f c2[2] = {};
  for (int k = 0; k < 8; ++k) {
    v16h a = afrag(&H[lrow * 264 + k * 32 + lhalf * 8]);
#pragma unroll
    for (int j = 0; j < 2; ++j) {
      int nT = wave * 2 + j;
      v16h b = *(const v16h*)&pW2[(size_t)((nT * 8 + k) * 32 + lane) * 16];
      c2[j] = wmma16(a, b, c2[j]);
    }
  }
#pragma unroll
  for (int j = 0; j < 2; ++j) {
    int ncol = (wave * 2 + j) * 16 + lrow;
    float bias = b2[ncol];
#pragma unroll
    for (int r = 0; r < 8; ++r) {
      int m    = r + lhalf * 8;
      int node = n0 + m;
      float val = c2[j][r] + bias;
      out[(size_t)node * ND + ncol] =
          (sDeg[m] > 0) ? val : nodeF[(size_t)node * ND + ncol];
    }
  }
}

// ---------------- passthrough edge features ----------------
__global__ void k_copy(const float4* __restrict__ in, float4* __restrict__ out, int n4) {
  int i = blockIdx.x * blockDim.x + threadIdx.x;
  if (i < n4) out[i] = in[i];
}

// ---------------- launch ----------------
extern "C" void kernel_launch(void* const* d_in, const int* in_sizes, int n_in,
                              void* d_out, int out_size, void* d_ws, size_t ws_size,
                              hipStream_t stream) {
  const float* nodeF = (const float*)d_in[0];
  const float* edgeF = (const float*)d_in[1];
  const int*   src   = (const int*)d_in[2];
  const int*   dst   = (const int*)d_in[3];
  const float* We1   = (const float*)d_in[4];
  const float* be1   = (const float*)d_in[5];
  const float* We2   = (const float*)d_in[6];
  const float* be2   = (const float*)d_in[7];
  const float* Wn1   = (const float*)d_in[8];
  const float* bn1   = (const float*)d_in[9];
  const float* Wn2   = (const float*)d_in[10];
  const float* bn2   = (const float*)d_in[11];

  // workspace layout (all 16B aligned)
  _Float16* pWe1 = (_Float16*)d_ws;        // 320x256 -> 81920 halfs
  _Float16* pWe2 = pWe1 + 81920;           // 256x64  -> 16384 halfs
  _Float16* pWn1 = pWe2 + 16384;           // 192x256 -> 49152 halfs
  _Float16* pWn2 = pWn1 + 49152;           // 256x128 -> 32768 halfs
  float* esum = (float*)(pWn2 + 32768);    // 50000*64 f32
  int*   deg  = (int*)(esum + (size_t)NN * ED);  // 50000 ints (contiguous after esum)

  // zero esum + deg (contiguous dword region, count divisible by 4)
  int zDwords = NN * ED + NN;
  int z4 = zDwords / 4;
  k_zero<<<(z4 + 255) / 256, 256, 0, stream>>>((int4*)esum, z4);

  // pack weights to f16 B-fragment layout
  k_pack<<<(81920 + 255) / 256, 256, 0, stream>>>(We1, pWe1, 10, 256, 81920);
  k_pack<<<(16384 + 255) / 256, 256, 0, stream>>>(We2, pWe2, 8, 64, 16384);
  k_pack<<<(49152 + 255) / 256, 256, 0, stream>>>(Wn1, pWn1, 6, 256, 49152);
  k_pack<<<(32768 + 255) / 256, 256, 0, stream>>>(Wn2, pWn2, 8, 128, 32768);

  k_deg<<<(NE + 255) / 256, 256, 0, stream>>>(dst, deg);

  k_edge<<<NE / 64, 128, 0, stream>>>(nodeF, edgeF, src, dst,
                                      pWe1, be1, pWe2, be2, esum);

  float* hOut = (float*)d_out;
  k_node<<<NN / 16, 128, 0, stream>>>(nodeF, esum, deg, pWn1, bn1, pWn2, bn2, hOut);

  int n4 = NE * ED / 4;  // 12.8M float4
  k_copy<<<(n4 + 255) / 256, 256, 0, stream>>>((const float4*)edgeF,
                                               (float4*)(hOut + (size_t)NN * ND), n4);
}
